// Encoder_Decoder_attention_fc_81621558493668
// MI455X (gfx1250) — matrix operand order repaired and checked by reading, not compiled, not run on hardware
//
#include <hip/hip_runtime.h>
#include <hip/hip_bf16.h>

// ---------------------------------------------------------------------------
// Encoder/decoder LSTM + attention for MI455X (gfx1250).
// Persistent per-batch-slice workgroups; recurrent GEMMs on v_wmma bf16;
// fp32 cell state / softmax; encoder states cached in L2 as bf16 (32 MB).
// ---------------------------------------------------------------------------

#define Bq 64
#define Tt 1024
#define Hh 256
#define G4 1024          // 4*H

typedef __attribute__((ext_vector_type(16))) __bf16 v16bf;
typedef __attribute__((ext_vector_type(8)))  float  v8f;

union Frag16 { v16bf v; uint4 q[2]; };

// dynamic-LDS partition (bytes)
#define OFF_GATES 0              // 16*1024 f32 = 64 KB (also attention scores)
#define OFF_C     (64*1024)      // 16*256  f32 = 16 KB
#define OFF_HBF   (80*1024)      // 16*256  bf16 = 8 KB
#define OFF_XS    (88*1024)      // 16 f32
#define OFF_RED   (88*1024+256)  // 16*64 f32 = 4 KB
#define OFF_ROWV  (88*1024+256+4096) // 16 f32
#define SMEM_BYTES (96*1024)

__device__ __forceinline__ float sigmoidf_(float x) { return 1.0f / (1.0f + __expf(-x)); }

__device__ __forceinline__ unsigned short f2bf(float f) {
    union { float f; unsigned u; } a; a.f = f;
    unsigned u = a.u;
    unsigned r = u + 0x7FFFu + ((u >> 16) & 1u);   // RNE
    return (unsigned short)(r >> 16);
}
__device__ __forceinline__ float bf2f(unsigned short h) {
    union { unsigned u; float f; } a; a.u = ((unsigned)h) << 16; return a.f;
}

__device__ __forceinline__ float dot8bf(uint4 a, uint4 b) {
    const unsigned* au = (const unsigned*)&a;
    const unsigned* bu = (const unsigned*)&b;
    float s = 0.f;
#pragma unroll
    for (int i = 0; i < 4; ++i) {
        s = fmaf(bf2f((unsigned short)(au[i] & 0xFFFFu)), bf2f((unsigned short)(bu[i] & 0xFFFFu)), s);
        s = fmaf(bf2f((unsigned short)(au[i] >> 16)),     bf2f((unsigned short)(bu[i] >> 16)),     s);
    }
    return s;
}

// ---------------------------------------------------------------------------
// One LSTM cell step for a 16-row batch slice, 1024 threads (32 wave32).
//   gates_raw = h @ W^T    via v_wmma_f32_16x16x32_bf16 (M=16,N=1024,K=256)
//   + bias + x*Wih, then fused activations + c/h update.
// On exit: hbf holds new h (bf16), cst new c (f32), gates[m*G4+hh] new h (f32).
// ---------------------------------------------------------------------------
__device__ __forceinline__ void lstm_step(const unsigned short* __restrict__ Wbf, // [G4][Hh] bf16 row-major
                                          const float* __restrict__ Wih,          // [G4]
                                          const float* __restrict__ bias,         // [G4]
                                          const float* xs,                        // LDS [16]
                                          unsigned short* hbf,                    // LDS [16][256]
                                          float* cst,                             // LDS [16][256]
                                          float* gates,                           // LDS [16][1024]
                                          int tid)
{
    // Opaque zero offset, re-materialized every (inlined) call: keeps the
    // weight loads non-loop-invariant across the caller's time loop (so LICM
    // can't hoist 128 VGPRs of weights and spill them to scratch — round 1),
    // while preserving kernarg pointer provenance so the accesses still lower
    // to global_load_b128 (not flat_load — round 2).
    int zoff = 0;
    asm volatile("" : "+s"(zoff));

    const int lane  = tid & 31;
    const int wave  = tid >> 5;
    const int row   = lane & 15;
    const int hi    = (lane >> 4) & 1;      // 0: lanes 0-15, 1: lanes 16-31
    const int n0    = (wave * 2)     * 16 + row;
    const int n1    = (wave * 2 + 1) * 16 + row;
    const unsigned short* w0 = Wbf + n0 * Hh + zoff;
    const unsigned short* w1 = Wbf + n1 * Hh + zoff;

    v8f acc0 = {}; v8f acc1 = {};
#pragma unroll
    for (int k0 = 0; k0 < Hh; k0 += 32) {
        // A fragment: 16-bit A 16x32 layout (lanes<16: K 0-7/16-23, lanes>=16: K 8-15/24-31)
        Frag16 a, b0, b1;
        const int ak = k0 + (hi ? 8 : 0);
        a.q[0] = *(const uint4*)&hbf[row * Hh + ak];
        a.q[1] = *(const uint4*)&hbf[row * Hh + ak + 16];
        // B fragment: 16-bit B 32x16 layout (lane=col, lanes<16: K 0-7/16-23, lanes>=16: K 8-15/24-31)
        const int bk = k0 + (hi ? 8 : 0);
        b0.q[0] = *(const uint4*)&w0[bk];
        b0.q[1] = *(const uint4*)&w0[bk + 16];
        b1.q[0] = *(const uint4*)&w1[bk];
        b1.q[1] = *(const uint4*)&w1[bk + 16];
        __builtin_prefetch(w0 + bk + 32, 0, 1);   // global_prefetch_b8 (next K slab)
        __builtin_prefetch(w1 + bk + 32, 0, 1);
        acc0 = __builtin_amdgcn_wmma_f32_16x16x32_bf16(false, a.v, false, b0.v, (short)0, acc0, false, false);
        acc1 = __builtin_amdgcn_wmma_f32_16x16x32_bf16(false, a.v, false, b1.v, (short)0, acc1, false, false);
    }

    // C/D layout: vgpr e, lane L -> m = e + 8*(L>=16), n = tile + (L&15)
    const int mb   = hi ? 8 : 0;
    const float wi0 = Wih[n0], wi1 = Wih[n1];
    const float bb0 = bias[n0], bb1 = bias[n1];
#pragma unroll
    for (int e = 0; e < 8; ++e) {
        const int m = mb + e;
        gates[m * G4 + n0] = acc0[e] + bb0 + xs[m] * wi0;
        gates[m * G4 + n1] = acc1[e] + bb1 + xs[m] * wi1;
    }
    __syncthreads();

    // fused activations + cell update (torch gate order i,f,g,o)
    for (int idx = tid; idx < 16 * Hh; idx += 1024) {
        const int m = idx >> 8, hh = idx & 255;
        const float iv = sigmoidf_(gates[m * G4 + hh]);
        const float fv = sigmoidf_(gates[m * G4 + hh + 256]);
        const float gv = tanhf    (gates[m * G4 + hh + 512]);
        const float ov = sigmoidf_(gates[m * G4 + hh + 768]);
        const float c  = fv * cst[idx] + iv * gv;
        cst[idx] = c;
        const float h = ov * tanhf(c);
        gates[m * G4 + hh] = h;       // stash fp32 h (i-slot is dead now)
        hbf[idx] = f2bf(h);
    }
    __syncthreads();
}

// ---------------------------------------------------------------------------
__global__ void prep_kernel(const float* __restrict__ encWhh, const float* __restrict__ decWhh,
                            const float* __restrict__ encbih, const float* __restrict__ encbhh,
                            const float* __restrict__ decbih, const float* __restrict__ decbhh,
                            unsigned short* __restrict__ encWbf, unsigned short* __restrict__ decWbf,
                            float* __restrict__ encBias, float* __restrict__ decBias)
{
    const int i = blockIdx.x * blockDim.x + threadIdx.x;
    const int n = G4 * Hh;
    for (int idx = i; idx < n; idx += gridDim.x * blockDim.x) {
        encWbf[idx] = f2bf(encWhh[idx]);
        decWbf[idx] = f2bf(decWhh[idx]);
    }
    if (i < G4) {
        encBias[i] = encbih[i] + encbhh[i];
        decBias[i] = decbih[i] + decbhh[i];
    }
}

// ---------------------------------------------------------------------------
__global__ void __launch_bounds__(1024)
encoder_kernel(const float* __restrict__ x,            // [B,T,1]
               const float* __restrict__ Wih,          // [G4]
               const unsigned short* __restrict__ Wbf, // [G4][Hh] bf16
               const float* __restrict__ bias,         // [G4]
               unsigned short* __restrict__ encSt,     // [B,T,H] bf16
               float* __restrict__ hlast)              // [B,H] f32
{
    extern __shared__ char smem[];
    float*          gates = (float*)(smem + OFF_GATES);
    float*          cst   = (float*)(smem + OFF_C);
    unsigned short* hbf   = (unsigned short*)(smem + OFF_HBF);
    float*          xs    = (float*)(smem + OFF_XS);

    const int tid = threadIdx.x;
    const int b0  = blockIdx.x * 16;

    for (int idx = tid; idx < 16 * Hh; idx += 1024) { cst[idx] = 0.f; hbf[idx] = 0; }
    __syncthreads();

    for (int t = 0; t < Tt; ++t) {
        if (tid < 16) xs[tid] = x[(b0 + tid) * Tt + t];
        __syncthreads();
        lstm_step(Wbf, Wih, bias, xs, hbf, cst, gates, tid);
        for (int idx = tid; idx < 16 * Hh; idx += 1024) {
            const int m = idx >> 8, hh = idx & 255;
            encSt[(size_t)((b0 + m) * Tt + t) * Hh + hh] = hbf[idx];
        }
        if (t == Tt - 1) {
            for (int idx = tid; idx < 16 * Hh; idx += 1024) {
                const int m = idx >> 8, hh = idx & 255;
                hlast[(b0 + m) * Hh + hh] = gates[m * G4 + hh];
            }
        }
        __syncthreads();
    }
}

// ---------------------------------------------------------------------------
__global__ void __launch_bounds__(1024)
decoder_kernel(const float* __restrict__ x,
               const float* __restrict__ Wih,
               const unsigned short* __restrict__ Wbf,
               const float* __restrict__ bias,
               const unsigned short* __restrict__ encSt, // [B,T,H] bf16 (L2-resident)
               const float* __restrict__ hlast,          // [B,H]
               const float* __restrict__ linW,           // [2H]
               const float* __restrict__ linb,           // [1]
               float* __restrict__ out)                  // [B,T,1]
{
    extern __shared__ char smem[];
    float*          gates = (float*)(smem + OFF_GATES);   // reused as scores/attn
    float*          cst   = (float*)(smem + OFF_C);
    unsigned short* hbf   = (unsigned short*)(smem + OFF_HBF);
    float*          xs    = (float*)(smem + OFF_XS);
    float*          red   = (float*)(smem + OFF_RED);     // [16][64]
    float*          rowv  = (float*)(smem + OFF_ROWV);    // [16]

    const int tid = threadIdx.x;
    const int b0  = blockIdx.x * 16;
    const float lb = linb[0];

    for (int idx = tid; idx < 16 * Hh; idx += 1024) {
        cst[idx] = 0.f;
        const int m = idx >> 8, hh = idx & 255;
        hbf[idx] = f2bf(hlast[(b0 + m) * Hh + hh]);
    }
    __syncthreads();

    for (int t = 0; t < Tt; ++t) {
        if (tid < 16) xs[tid] = x[(b0 + tid) * Tt + t];
        __syncthreads();
        lstm_step(Wbf, Wih, bias, xs, hbf, cst, gates, tid);   // ends with barrier

        // ---- attention scores: sc[m][t'] = <enc[b0+m][t'], hx[m]> ----
        {
            const int m  = tid & 15;
            const int tg = tid >> 4;                 // 0..63
#pragma unroll 1
            for (int i = 0; i < 16; ++i) {
                const int tp = tg + 64 * i;
                const unsigned short* er = encSt + (size_t)((b0 + m) * Tt + tp) * Hh;
                float s = 0.f;
#pragma unroll 4
                for (int k = 0; k < Hh; k += 8) {
                    const uint4 ev = *(const uint4*)&er[k];
                    const uint4 hv = *(const uint4*)&hbf[m * Hh + k];
                    s += dot8bf(ev, hv);
                }
                gates[m * G4 + tp] = s;
            }
        }
        __syncthreads();

        // ---- softmax over t' (rows of 1024, 64 threads per batch row) ----
        {
            const int m = tid >> 6;                  // 0..15
            const int j = tid & 63;
            float mx = -1e30f;
#pragma unroll
            for (int i = 0; i < 16; ++i) mx = fmaxf(mx, gates[m * G4 + j + 64 * i]);
            red[m * 64 + j] = mx;
            __syncthreads();
            if (j == 0) {
                float v = red[m * 64];
                for (int k = 1; k < 64; ++k) v = fmaxf(v, red[m * 64 + k]);
                rowv[m] = v;
            }
            __syncthreads();
            const float rm = rowv[m];
            float sum = 0.f;
#pragma unroll
            for (int i = 0; i < 16; ++i) {
                const float e = __expf(gates[m * G4 + j + 64 * i] - rm);
                gates[m * G4 + j + 64 * i] = e;      // unnormalized weights
                sum += e;
            }
            red[m * 64 + j] = sum;
            __syncthreads();
            if (j == 0) {
                float v = 0.f;
                for (int k = 0; k < 64; ++k) v += red[m * 64 + k];
                rowv[m] = v;                          // row normalizer
            }
            __syncthreads();
        }

        // ---- context + linear head:  out = [ctx, hx] @ linW^T + b ----
        {
            const int m  = tid >> 6;
            const int j  = tid & 63;
            const int h0 = j * 4;
            float a0 = 0.f, a1 = 0.f, a2 = 0.f, a3 = 0.f;
            const unsigned short* eb = encSt + (size_t)(b0 + m) * Tt * Hh + h0;
#pragma unroll 4
            for (int tp = 0; tp < Tt; ++tp) {
                const float w = gates[m * G4 + tp];
                const uint2 ev = *(const uint2*)&eb[(size_t)tp * Hh];
                a0 = fmaf(w, bf2f((unsigned short)(ev.x & 0xFFFFu)), a0);
                a1 = fmaf(w, bf2f((unsigned short)(ev.x >> 16)),     a1);
                a2 = fmaf(w, bf2f((unsigned short)(ev.y & 0xFFFFu)), a2);
                a3 = fmaf(w, bf2f((unsigned short)(ev.y >> 16)),     a3);
            }
            const float inv = 1.0f / rowv[m];
            float p = 0.f;
            p = fmaf(linW[h0 + 0], a0 * inv, p);
            p = fmaf(linW[h0 + 1], a1 * inv, p);
            p = fmaf(linW[h0 + 2], a2 * inv, p);
            p = fmaf(linW[h0 + 3], a3 * inv, p);
#pragma unroll
            for (int q = 0; q < 4; ++q)
                p = fmaf(linW[Hh + h0 + q], bf2f(hbf[m * Hh + h0 + q]), p);
            red[m * 64 + j] = p;
            __syncthreads();
            if (j == 0) {
                float v = 0.f;
                for (int k = 0; k < 64; ++k) v += red[m * 64 + k];
                out[(b0 + m) * Tt + t] = v + lb;
            }
            __syncthreads();
        }
    }
}

// ---------------------------------------------------------------------------
extern "C" void kernel_launch(void* const* d_in, const int* in_sizes, int n_in,
                              void* d_out, int out_size, void* d_ws, size_t ws_size,
                              hipStream_t stream)
{
    const float* x      = (const float*)d_in[0];
    const float* eWih   = (const float*)d_in[1];
    const float* eWhh   = (const float*)d_in[2];
    const float* ebih   = (const float*)d_in[3];
    const float* ebhh   = (const float*)d_in[4];
    const float* dWih   = (const float*)d_in[5];
    const float* dWhh   = (const float*)d_in[6];
    const float* dbih   = (const float*)d_in[7];
    const float* dbhh   = (const float*)d_in[8];
    const float* linW   = (const float*)d_in[9];
    const float* linb   = (const float*)d_in[10];
    float* out = (float*)d_out;

    char* ws = (char*)d_ws;
    size_t off = 0;
    unsigned short* encWbf = (unsigned short*)(ws + off); off += (size_t)G4 * Hh * 2;
    unsigned short* decWbf = (unsigned short*)(ws + off); off += (size_t)G4 * Hh * 2;
    float* encBias = (float*)(ws + off); off += G4 * 4;
    float* decBias = (float*)(ws + off); off += G4 * 4;
    float* hlast   = (float*)(ws + off); off += (size_t)Bq * Hh * 4;
    off = (off + 255) & ~(size_t)255;
    unsigned short* encSt = (unsigned short*)(ws + off); off += (size_t)Bq * Tt * Hh * 2; // 32 MB

    (void)in_sizes; (void)n_in; (void)out_size; (void)ws_size;

    hipFuncSetAttribute((const void*)encoder_kernel,
                        hipFuncAttributeMaxDynamicSharedMemorySize, SMEM_BYTES);
    hipFuncSetAttribute((const void*)decoder_kernel,
                        hipFuncAttributeMaxDynamicSharedMemorySize, SMEM_BYTES);

    prep_kernel<<<512, 512, 0, stream>>>(eWhh, dWhh, ebih, ebhh, dbih, dbhh,
                                         encWbf, decWbf, encBias, decBias);
    encoder_kernel<<<Bq / 16, 1024, SMEM_BYTES, stream>>>(x, eWih, encWbf, encBias, encSt, hlast);
    decoder_kernel<<<Bq / 16, 1024, SMEM_BYTES, stream>>>(x, dWih, decWbf, decBias, encSt, hlast,
                                                          linW, linb, out);
}
